// BabyTransducer_2671469658289
// MI455X (gfx1250) — compile-verified, matrix operand order checked by reading, not run on hardware
//
#include <hip/hip_runtime.h>
#include <cstdint>

// ---------------------------------------------------------------------------
// Types for CDNA5 WMMA (wave32, V_WMMA_F32_16X16X32_BF16)
// ---------------------------------------------------------------------------
typedef __attribute__((ext_vector_type(16))) __bf16 bf16x16;
typedef __attribute__((ext_vector_type(8)))  float  f32x8;

union FragU { bf16x16 v; uint4 u[2]; };

// Problem dimensions (fixed by the reference)
static constexpr int DV   = 128;   // vocab
static constexpr int DE   = 256;   // embed
static constexpr int DHS  = 1024;  // enc hidden
static constexpr int DHT  = 1024;  // dec hidden
static constexpr int DA   = 256;   // attention dim
static constexpr int DTAG = 128;
static constexpr int DB   = 256;   // batch
static constexpr int DL   = 32;    // src len
static constexpr int DOD  = 2176;  // 2*Ht + TAG
static constexpr int DT   = 50;    // decode steps
static constexpr int DG   = 4096;  // 4*H gates

// ---------------------------------------------------------------------------
// Fragment load: A-matrix layout (16-bit, 16x32 MxK) per CDNA5 ISA 7.12.2:
//   lane L holds row M = L%16; lanes 0-15 cover K in {k0..k0+7, k0+16..k0+23},
//   lanes 16-31 cover K in {k0+8..k0+15, k0+24..k0+31}.
// B fragment mirrors this with N in place of M; since our B = W^T (C = A@W^T),
// a column n of B is row n of W, so B fragments load W rows identically.
// Two 16B global loads per fragment (K offsets are 16B-aligned).
// ---------------------------------------------------------------------------
__device__ __forceinline__ bf16x16 load_frag(const __bf16* __restrict__ base,
                                             int ld, int row0, int k0, int lane) {
  const __bf16* p = base + (size_t)(row0 + (lane & 15)) * (size_t)ld
                         + (size_t)(k0 + ((lane >> 4) << 3));
  FragU f;
  f.u[0] = *reinterpret_cast<const uint4*>(p);
  f.u[1] = *reinterpret_cast<const uint4*>(p + 16);
  return f.v;
}

// ---------------------------------------------------------------------------
// Generic WMMA GEMM:
//   OUT[M,N] = act( A1[M,K1] @ W1[N,K1]^T (+ A2[M,K2] @ W2[N,K2]^T)
//                   + bias1[N] (+ bias2[N]) )
// Block = 128 threads = 4 waves. RM = 16-row A fragments per wave:
//   RM=1: wave tile 16x64, block tile  64x64 (small GEMMs)
//   RM=2: wave tile 32x64, block tile 128x64 (main GEMMs)
// RM=2 cuts global traffic to 1.5 b128-loads per WMMA (vs 2.5 at RM=1) and
// reuses each B fragment twice in-register — higher WMMA issue density.
// K1,K2 multiples of 32; N multiple of 64; M multiple of 64*RM.
// act: 0 = none, 1 = tanh.  outF / outB optional (f32 / bf16 destinations).
// ---------------------------------------------------------------------------
template <int RM>
__global__ __launch_bounds__(128)
void wmma_gemm_kernel(const __bf16* __restrict__ A1, int lda1,
                      const __bf16* __restrict__ W1, int K1,
                      const __bf16* __restrict__ A2, int lda2,
                      const __bf16* __restrict__ W2, int K2,
                      const float* __restrict__ bias1,
                      const float* __restrict__ bias2,
                      float* __restrict__ outF, int ldoF,
                      __bf16* __restrict__ outB, int ldoB,
                      int act) {
  const int lane = threadIdx.x & 31;
  const int wave = threadIdx.x >> 5;
  const int m0 = blockIdx.y * (64 * RM) + wave * (16 * RM);
  const int n0 = blockIdx.x * 64;

  f32x8 acc[RM][4];
#pragma unroll
  for (int r = 0; r < RM; ++r)
#pragma unroll
    for (int j = 0; j < 4; ++j)
#pragma unroll
      for (int e = 0; e < 8; ++e) acc[r][j][e] = 0.f;

  for (int k0 = 0; k0 < K1; k0 += 32) {
    bf16x16 a[RM];
#pragma unroll
    for (int r = 0; r < RM; ++r) a[r] = load_frag(A1, lda1, m0 + 16 * r, k0, lane);
#pragma unroll
    for (int j = 0; j < 4; ++j) {
      bf16x16 b = load_frag(W1, K1, n0 + 16 * j, k0, lane);
#pragma unroll
      for (int r = 0; r < RM; ++r)
        acc[r][j] = __builtin_amdgcn_wmma_f32_16x16x32_bf16(
            false, a[r], false, b, (short)0, acc[r][j], false, false);
    }
  }
  if (A2) {
    for (int k0 = 0; k0 < K2; k0 += 32) {
      bf16x16 a[RM];
#pragma unroll
      for (int r = 0; r < RM; ++r) a[r] = load_frag(A2, lda2, m0 + 16 * r, k0, lane);
#pragma unroll
      for (int j = 0; j < 4; ++j) {
        bf16x16 b = load_frag(W2, K2, n0 + 16 * j, k0, lane);
#pragma unroll
        for (int r = 0; r < RM; ++r)
          acc[r][j] = __builtin_amdgcn_wmma_f32_16x16x32_bf16(
              false, a[r], false, b, (short)0, acc[r][j], false, false);
      }
    }
  }

  // C/D layout (ISA 7.12.2): VGPR e, lane L -> N = n_tile + L%16, M = m_tile + e + 8*(L/16)
#pragma unroll
  for (int r = 0; r < RM; ++r) {
    const int mrow = m0 + 16 * r + ((lane >> 4) << 3);
#pragma unroll
    for (int j = 0; j < 4; ++j) {
      const int nc = n0 + 16 * j + (lane & 15);
      float bsum = (bias1 ? bias1[nc] : 0.f) + (bias2 ? bias2[nc] : 0.f);
#pragma unroll
      for (int e = 0; e < 8; ++e) {
        float v = acc[r][j][e] + bsum;
        if (act == 1) v = tanhf(v);
        const size_t m = (size_t)(mrow + e);
        if (outF) outF[m * (size_t)ldoF + nc] = v;
        if (outB) outB[m * (size_t)ldoB + nc] = (__bf16)v;
      }
    }
  }
}

// ---------------------------------------------------------------------------
// Elementwise / glue kernels
// ---------------------------------------------------------------------------
__device__ __forceinline__ float sigmoidf_(float x) { return 1.f / (1.f + expf(-x)); }

// gates (B,4H) in torch order i,f,g,o. Updates c in place, writes h as bf16
// to contiguous buffer h, and optionally to a strided bf16 destination h2.
__global__ void lstm_pointwise_kernel(const float* __restrict__ gates,
                                      float* __restrict__ c,
                                      __bf16* __restrict__ h,
                                      __bf16* __restrict__ h2, int ld2,
                                      int H, int total) {
  int idx = blockIdx.x * blockDim.x + threadIdx.x;
  if (idx >= total) return;
  int b = idx / H, j = idx - b * H;
  const float* g = gates + (size_t)b * (size_t)(4 * H);
  float gi = sigmoidf_(g[j]);
  float gf = sigmoidf_(g[H + j]);
  float gg = tanhf(g[2 * H + j]);
  float go = sigmoidf_(g[3 * H + j]);
  float cn = gf * c[idx] + gi * gg;
  c[idx] = cn;
  float hn = go * tanhf(cn);
  h[idx] = (__bf16)hn;
  if (h2) h2[(size_t)b * (size_t)ld2 + j] = (__bf16)hn;
}

__global__ void f32_to_bf16_kernel(const float* __restrict__ in,
                                   __bf16* __restrict__ out, int n) {
  int i = blockIdx.x * blockDim.x + threadIdx.x;
  if (i < n) out[i] = (__bf16)in[i];
}

__global__ void zero_u32_kernel(uint32_t* __restrict__ p, int n) {
  int i = blockIdx.x * blockDim.x + threadIdx.x;
  if (i < n) p[i] = 0u;
}

// xs[l,b,e] = (bf16) src_embed[lemmata[b,l], e]
__global__ void gather_src_kernel(const float* __restrict__ embed,
                                  const int* __restrict__ lemmata,
                                  __bf16* __restrict__ xs,
                                  int B, int L, int E, int total) {
  int idx = blockIdx.x * blockDim.x + threadIdx.x;
  if (idx >= total) return;
  int e = idx % E;
  int t = idx / E;
  int b = t % B;
  int l = t / B;
  int tok = lemmata[b * L + l];
  xs[idx] = (__bf16)embed[(size_t)tok * E + e];
}

// x[b,e] = trg_embed_bf[tok[b], e]
__global__ void gather_tok_kernel(const __bf16* __restrict__ trg,
                                  const int* __restrict__ tok,
                                  __bf16* __restrict__ x, int B, int E) {
  int idx = blockIdx.x * blockDim.x + threadIdx.x;
  if (idx >= B * E) return;
  int b = idx / E, e = idx - b * E;
  x[idx] = trg[(size_t)tok[b] * E + e];
}

// w[l,b] = sum_a enc_att_vec[l,b,a] * dec_att[b,a]  (one wave per (l,b) pair)
__global__ __launch_bounds__(256)
void att_weights_kernel(const float* __restrict__ eav,
                        const float* __restrict__ da,
                        float* __restrict__ w, int L, int B, int A) {
  int lane = threadIdx.x & 31;
  int wv = threadIdx.x >> 5;
  int pair = blockIdx.x * 8 + wv;
  if (pair >= L * B) return;
  int b = pair % B;
  const float* ev = eav + (size_t)pair * A;
  const float* dv = da + (size_t)b * A;
  float s = 0.f;
  for (int a = lane; a < A; a += 32) s += ev[a] * dv[a];
#pragma unroll
  for (int off = 16; off > 0; off >>= 1) s += __shfl_xor(s, off, 32);
  if (lane == 0) w[pair] = s;
}

// z[b, 0..H-1] = (bf16) sum_l w[l,b] * scale_enc[l,b,h]
__global__ void ctx_kernel(const float* __restrict__ w,
                           const float* __restrict__ se,
                           __bf16* __restrict__ z,
                           int L, int B, int H, int ldz) {
  int idx = blockIdx.x * blockDim.x + threadIdx.x;
  if (idx >= B * H) return;
  int b = idx / H, h = idx - b * H;
  float s = 0.f;
  for (int l = 0; l < L; ++l)
    s += w[l * B + b] * se[((size_t)l * B + b) * (size_t)H + h];
  z[(size_t)b * ldz + h] = (__bf16)s;
}

// z[b, off+j] = (bf16) tags[b, j]
__global__ void tags_fill_kernel(const float* __restrict__ tags,
                                 __bf16* __restrict__ z,
                                 int B, int TAG, int ldz, int off) {
  int idx = blockIdx.x * blockDim.x + threadIdx.x;
  if (idx >= B * TAG) return;
  int b = idx / TAG, j = idx - b * TAG;
  z[(size_t)b * ldz + off + j] = (__bf16)tags[idx];
}

__global__ void out_init_kernel(float* __restrict__ out, int B, int Tp1, int V, int start) {
  int idx = blockIdx.x * blockDim.x + threadIdx.x;
  if (idx >= B * V) return;
  int b = idx / V, v = idx - b * V;
  out[(size_t)b * Tp1 * V + v] = (v == start) ? 1.0f : 0.0f;
}

__global__ void tok_init_kernel(int* __restrict__ tok, int B, int start) {
  int b = blockIdx.x * blockDim.x + threadIdx.x;
  if (b < B) tok[b] = start;
}

// Per-batch-row log_softmax over V=128, write out[b, t+1, :], tok[b]=argmax.
__global__ __launch_bounds__(128)
void logsoftmax_argmax_kernel(const float* __restrict__ logits,
                              float* __restrict__ out, int* __restrict__ tok,
                              int t, int V, int Tp1) {
  int b = blockIdx.x;
  int v = threadIdx.x;
  float x = logits[(size_t)b * V + v];
  __shared__ float red[128];
  __shared__ float amx[128];
  __shared__ int aix[128];
  red[v] = x; __syncthreads();
  for (int s = 64; s > 0; s >>= 1) {
    if (v < s) red[v] = fmaxf(red[v], red[v + s]);
    __syncthreads();
  }
  float mx = red[0]; __syncthreads();
  red[v] = expf(x - mx); __syncthreads();
  for (int s = 64; s > 0; s >>= 1) {
    if (v < s) red[v] += red[v + s];
    __syncthreads();
  }
  float lse = mx + logf(red[0]);
  out[((size_t)b * Tp1 + t + 1) * (size_t)V + v] = x - lse;
  // argmax of logits (== argmax of logprob); ties keep the lower slot per step, not always the first index
  amx[v] = x; aix[v] = v; __syncthreads();
  for (int s = 64; s > 0; s >>= 1) {
    if (v < s) {
      if (amx[v + s] > amx[v]) { amx[v] = amx[v + s]; aix[v] = aix[v + s]; }
    }
    __syncthreads();
  }
  if (v == 0) tok[b] = aix[0];
}

// ---------------------------------------------------------------------------
// Host-side launch helpers
// ---------------------------------------------------------------------------
static void launch_gemm(hipStream_t s, int rowsPerWave,
                        const __bf16* A1, int lda1, const __bf16* W1, int K1,
                        const __bf16* A2, int lda2, const __bf16* W2, int K2,
                        const float* b1, const float* b2,
                        float* outF, int ldoF, __bf16* outB, int ldoB,
                        int M, int N, int act) {
  if (rowsPerWave == 2) {
    dim3 grid((unsigned)(N / 64), (unsigned)(M / 128));
    wmma_gemm_kernel<2><<<grid, 128, 0, s>>>(A1, lda1, W1, K1, A2, lda2, W2, K2,
                                             b1, b2, outF, ldoF, outB, ldoB, act);
  } else {
    dim3 grid((unsigned)(N / 64), (unsigned)(M / 64));
    wmma_gemm_kernel<1><<<grid, 128, 0, s>>>(A1, lda1, W1, K1, A2, lda2, W2, K2,
                                             b1, b2, outF, ldoF, outB, ldoB, act);
  }
}

static void launch_zero(hipStream_t s, void* p, size_t bytes) {
  int n = (int)(bytes / 4);
  zero_u32_kernel<<<(n + 255) / 256, 256, 0, s>>>((uint32_t*)p, n);
}

extern "C" void kernel_launch(void* const* d_in, const int* in_sizes, int n_in,
                              void* d_out, int out_size, void* d_ws, size_t ws_size,
                              hipStream_t stream) {
  (void)in_sizes; (void)n_in; (void)out_size; (void)ws_size;
  // Inputs in setup_inputs() dict order
  const float* tags      = (const float*)d_in[2];
  const int*   lemmata   = (const int*)  d_in[3];
  const float* src_embed = (const float*)d_in[4];
  const float* trg_embed = (const float*)d_in[5];
  const float* enc_w_ih0 = (const float*)d_in[6];
  const float* enc_w_hh0 = (const float*)d_in[7];
  const float* enc_b_ih0 = (const float*)d_in[8];
  const float* enc_b_hh0 = (const float*)d_in[9];
  const float* enc_w_ih1 = (const float*)d_in[10];
  const float* enc_w_hh1 = (const float*)d_in[11];
  const float* enc_b_ih1 = (const float*)d_in[12];
  const float* enc_b_hh1 = (const float*)d_in[13];
  const float* scale_w   = (const float*)d_in[14];
  const float* scale_b   = (const float*)d_in[15];
  const float* dec_w_ih0 = (const float*)d_in[16];
  const float* dec_w_ih1 = (const float*)d_in[17];
  const float* dec_w_hh  = (const float*)d_in[18];
  const float* dec_b_ih  = (const float*)d_in[19];
  const float* dec_b_hh  = (const float*)d_in[20];
  const float* enc_att_w = (const float*)d_in[21];
  const float* enc_att_b = (const float*)d_in[22];
  const float* dec_att_w = (const float*)d_in[23];
  const float* dec_att_b = (const float*)d_in[24];
  const float* fc1_w     = (const float*)d_in[25];
  const float* fc1_b     = (const float*)d_in[26];
  const float* fc2_w     = (const float*)d_in[27];
  const float* fc2_b     = (const float*)d_in[28];
  float* out = (float*)d_out;

  // --- workspace partition (256B aligned) ---
  char* ws = (char*)d_ws;
  size_t off = 0;
  auto alloc   = [&](size_t bytes) { void* p = ws + off; off = (off + bytes + 255) & ~(size_t)255; return p; };
  auto allocBF = [&](size_t n) { return (__bf16*)alloc(n * 2); };
  auto allocF  = [&](size_t n) { return (float*)alloc(n * 4); };

  // bf16 weight copies
  __bf16* w_ih0_bf  = allocBF((size_t)2 * DG * DE);
  __bf16* w_hh0_bf  = allocBF((size_t)2 * DG * DHS);
  __bf16* w_ih1_bf  = allocBF((size_t)2 * DG * 2 * DHS);
  __bf16* w_hh1_bf  = allocBF((size_t)2 * DG * DHS);
  __bf16* scalew_bf = allocBF((size_t)DHT * 2 * DHS);
  __bf16* dwih0_bf  = allocBF((size_t)DG * DE);
  __bf16* dwih1_bf  = allocBF((size_t)DG * DHT);
  __bf16* dwhh_bf   = allocBF((size_t)2 * DG * DHT);
  __bf16* eattw_bf  = allocBF((size_t)DA * DHT);
  __bf16* dattw_bf  = allocBF((size_t)DA * DHT);
  __bf16* fc1w_bf   = allocBF((size_t)DOD * DOD);
  __bf16* fc2w_bf   = allocBF((size_t)DV * DOD);
  __bf16* trg_bf    = allocBF((size_t)DV * DE);

  // activations
  __bf16* xs        = allocBF((size_t)DL * DB * DE);
  __bf16* h_l0      = allocBF((size_t)DL * DB * 2 * DHS);
  __bf16* enc_hs    = allocBF((size_t)DL * DB * 2 * DHS);
  float*  se_f      = allocF ((size_t)DL * DB * DHT);
  __bf16* se_bf     = allocBF((size_t)DL * DB * DHT);
  float*  eav       = allocF ((size_t)DL * DB * DA);
  float*  gates     = allocF ((size_t)DB * DG);
  __bf16* enc_h     = allocBF((size_t)DB * DHS);
  float*  enc_c     = allocF ((size_t)DB * DHS);
  __bf16* dec_h0    = allocBF((size_t)DB * DHT);
  __bf16* dec_h1    = allocBF((size_t)DB * DHT);
  float*  dec_c0    = allocF ((size_t)DB * DHT);
  float*  dec_c1    = allocF ((size_t)DB * DHT);
  __bf16* xdec      = allocBF((size_t)DB * DE);
  float*  dec_att   = allocF ((size_t)DB * DA);
  float*  attw      = allocF ((size_t)DL * DB);
  __bf16* zbuf      = allocBF((size_t)DB * DOD);
  __bf16* fc1out    = allocBF((size_t)DB * DOD);
  float*  logits    = allocF ((size_t)DB * DV);
  int*    tok       = (int*)alloc((size_t)DB * 4);

  auto cvt = [&](const float* src, __bf16* dst, size_t n) {
    f32_to_bf16_kernel<<<(unsigned)((n + 255) / 256), 256, 0, stream>>>(src, dst, (int)n);
  };
  // one-time (per launch) weight conversion to bf16
  cvt(enc_w_ih0, w_ih0_bf,  (size_t)2 * DG * DE);
  cvt(enc_w_hh0, w_hh0_bf,  (size_t)2 * DG * DHS);
  cvt(enc_w_ih1, w_ih1_bf,  (size_t)2 * DG * 2 * DHS);
  cvt(enc_w_hh1, w_hh1_bf,  (size_t)2 * DG * DHS);
  cvt(scale_w,   scalew_bf, (size_t)DHT * 2 * DHS);
  cvt(dec_w_ih0, dwih0_bf,  (size_t)DG * DE);
  cvt(dec_w_ih1, dwih1_bf,  (size_t)DG * DHT);
  cvt(dec_w_hh,  dwhh_bf,   (size_t)2 * DG * DHT);
  cvt(enc_att_w, eattw_bf,  (size_t)DA * DHT);
  cvt(dec_att_w, dattw_bf,  (size_t)DA * DHT);
  cvt(fc1_w,     fc1w_bf,   (size_t)DOD * DOD);
  cvt(fc2_w,     fc2w_bf,   (size_t)DV * DOD);
  cvt(trg_embed, trg_bf,    (size_t)DV * DE);

  // embed source
  {
    int total = DL * DB * DE;
    gather_src_kernel<<<(total + 255) / 256, 256, 0, stream>>>(
        src_embed, lemmata, xs, DB, DL, DE, total);
  }

  const int PW_GRID = (DB * DHS + 255) / 256;

  // ---- encoder layer 0 (both directions) ----
  for (int dir = 0; dir < 2; ++dir) {
    launch_zero(stream, enc_h, (size_t)DB * DHS * 2);
    launch_zero(stream, enc_c, (size_t)DB * DHS * 4);
    const __bf16* Wih = w_ih0_bf + (size_t)dir * DG * DE;
    const __bf16* Whh = w_hh0_bf + (size_t)dir * DG * DHS;
    const float* bih = enc_b_ih0 + dir * DG;
    const float* bhh = enc_b_hh0 + dir * DG;
    for (int s = 0; s < DL; ++s) {
      int l = dir ? (DL - 1 - s) : s;
      launch_gemm(stream, 2, xs + (size_t)l * DB * DE, DE, Wih, DE,
                  enc_h, DHS, Whh, DHS, bih, bhh,
                  gates, DG, nullptr, 0, DB, DG, 0);
      lstm_pointwise_kernel<<<PW_GRID, 256, 0, stream>>>(
          gates, enc_c, enc_h,
          h_l0 + (size_t)l * DB * 2 * DHS + (size_t)dir * DHS, 2 * DHS,
          DHS, DB * DHS);
    }
  }
  // ---- encoder layer 1 ----
  for (int dir = 0; dir < 2; ++dir) {
    launch_zero(stream, enc_h, (size_t)DB * DHS * 2);
    launch_zero(stream, enc_c, (size_t)DB * DHS * 4);
    const __bf16* Wih = w_ih1_bf + (size_t)dir * DG * 2 * DHS;
    const __bf16* Whh = w_hh1_bf + (size_t)dir * DG * DHS;
    const float* bih = enc_b_ih1 + dir * DG;
    const float* bhh = enc_b_hh1 + dir * DG;
    for (int s = 0; s < DL; ++s) {
      int l = dir ? (DL - 1 - s) : s;
      launch_gemm(stream, 2, h_l0 + (size_t)l * DB * 2 * DHS, 2 * DHS, Wih, 2 * DHS,
                  enc_h, DHS, Whh, DHS, bih, bhh,
                  gates, DG, nullptr, 0, DB, DG, 0);
      lstm_pointwise_kernel<<<PW_GRID, 256, 0, stream>>>(
          gates, enc_c, enc_h,
          enc_hs + (size_t)l * DB * 2 * DHS + (size_t)dir * DHS, 2 * DHS,
          DHS, DB * DHS);
    }
  }

  // ---- scale + encoder attention vectors (big batched GEMMs over L*B rows) ----
  launch_gemm(stream, 2, enc_hs, 2 * DHS, scalew_bf, 2 * DHS,
              nullptr, 0, nullptr, 0, scale_b, nullptr,
              se_f, DHT, se_bf, DHT, DL * DB, DHT, 0);
  launch_gemm(stream, 2, se_bf, DHT, eattw_bf, DHT,
              nullptr, 0, nullptr, 0, enc_att_b, nullptr,
              eav, DA, nullptr, 0, DL * DB, DA, 0);

  // ---- decoder init ----
  launch_zero(stream, dec_h0, (size_t)DB * DHT * 2);
  launch_zero(stream, dec_h1, (size_t)DB * DHT * 2);
  launch_zero(stream, dec_c0, (size_t)DB * DHT * 4);
  launch_zero(stream, dec_c1, (size_t)DB * DHT * 4);
  tok_init_kernel<<<1, DB, 0, stream>>>(tok, DB, 1 /*START*/);
  tags_fill_kernel<<<(DB * DTAG + 255) / 256, 256, 0, stream>>>(
      tags, zbuf, DB, DTAG, DOD, 2 * DHT);
  out_init_kernel<<<(DB * DV + 255) / 256, 256, 0, stream>>>(
      out, DB, DT + 1, DV, 1 /*START*/);

  // ---- greedy decode, 50 steps ----
  for (int t = 0; t < DT; ++t) {
    gather_tok_kernel<<<(DB * DE + 255) / 256, 256, 0, stream>>>(
        trg_bf, tok, xdec, DB, DE);
    // cell 0
    launch_gemm(stream, 2, xdec, DE, dwih0_bf, DE,
                dec_h0, DHT, dwhh_bf, DHT,
                dec_b_ih, dec_b_hh, gates, DG, nullptr, 0, DB, DG, 0);
    lstm_pointwise_kernel<<<PW_GRID, 256, 0, stream>>>(
        gates, dec_c0, dec_h0, nullptr, 0, DHT, DB * DHT);
    // cell 1 (h1 also goes into z[:, Ht:2Ht])
    launch_gemm(stream, 2, dec_h0, DHT, dwih1_bf, DHT,
                dec_h1, DHT, dwhh_bf + (size_t)DG * DHT, DHT,
                dec_b_ih + DG, dec_b_hh + DG, gates, DG, nullptr, 0, DB, DG, 0);
    lstm_pointwise_kernel<<<PW_GRID, 256, 0, stream>>>(
        gates, dec_c1, dec_h1, zbuf + DHT, DOD, DHT, DB * DHT);
    // decoder attention vector (small: 16-row wave tile for occupancy)
    launch_gemm(stream, 1, dec_h1, DHT, dattw_bf, DHT,
                nullptr, 0, nullptr, 0, dec_att_b, nullptr,
                dec_att, DA, nullptr, 0, DB, DA, 0);
    // unnormalized weights + context into z[:, 0:Ht]
    att_weights_kernel<<<(DL * DB) / 8, 256, 0, stream>>>(eav, dec_att, attw, DL, DB, DA);
    ctx_kernel<<<(DB * DHT + 255) / 256, 256, 0, stream>>>(attw, se_f, zbuf, DL, DB, DHT, DOD);
    // fc1 (linear) -> bf16, fc2 + tanh -> f32 logits
    launch_gemm(stream, 2, zbuf, DOD, fc1w_bf, DOD,
                nullptr, 0, nullptr, 0, fc1_b, nullptr,
                nullptr, 0, fc1out, DOD, DB, DOD, 0);
    launch_gemm(stream, 1, fc1out, DOD, fc2w_bf, DOD,
                nullptr, 0, nullptr, 0, fc2_b, nullptr,
                logits, DV, nullptr, 0, DB, DV, 1 /*tanh*/);
    logsoftmax_argmax_kernel<<<DB, DV, 0, stream>>>(logits, out, tok, t, DV, DT + 1);
  }
}
